// Conv_6949257085179
// MI455X (gfx1250) — 8-bit matrix operand order repaired and checked by reading, compiled, not run on hardware
//
#include <hip/hip_runtime.h>

typedef __attribute__((ext_vector_type(16))) int   v16i;
typedef __attribute__((ext_vector_type(8)))  float v8f;

struct __align__(16) U128 { unsigned long long lo, hi; };

#define NB   32
#define C    256
#define HW   56
#define OC   256
#define HP   58
#define PIX  (HP*HP)
#define SP   (HW*HW)
#define M_TOT (NB*SP)
#define XP_BYTES ((long long)NB*PIX*C)
#define WP_BYTES ((long long)9*OC*C)

__device__ __forceinline__ unsigned char bin8(float v) {
    return (v <= 0.0f) ? (unsigned char)0xB8 : (unsigned char)0x38;
}

__global__ __launch_bounds__(256) void pack_x(const float* __restrict__ x,
                                              unsigned char* __restrict__ xp) {
    long long i = (long long)blockIdx.x * 256 + threadIdx.x;
    if (i >= (long long)NB*C*SP) return;
    int w = (int)(i % HW); long long t = i / HW;
    int h = (int)(t % HW); t /= HW;
    int c = (int)(t % C);  int n = (int)(t / C);
    unsigned char b = bin8(x[i]);
    xp[(((long long)n*HP + (h+1))*HP + (w+1))*C + c] = b;
}

__global__ __launch_bounds__(256) void pack_w(const float* __restrict__ wgt,
                                              unsigned char* __restrict__ wp) {
    int i = blockIdx.x * 256 + threadIdx.x;
    if (i >= 9*OC*C) return;
    int kw = i % 3; int t = i / 3;
    int kh = t % 3; t /= 3;
    int ic = t % C; int oc = t / C;
    wp[(((kh*3 + kw)*OC + oc))*(long long)C + ic] = bin8(wgt[i]);
}

__global__ __launch_bounds__(256) void bconv_wmma(const unsigned char* __restrict__ xp,
                                                  const unsigned char* __restrict__ wp,
                                                  float* __restrict__ out) {
    const int lane  = threadIdx.x & 31;
    const int wave  = threadIdx.x >> 5;
    const int waveM = wave & 1;
    const int waveN = wave >> 1;
    const int m0 = blockIdx.x * 64 + waveM * 32;
    const int n0 = waveN * 64;

    const int half = lane >> 4;

    long long pixA[2];
    #pragma unroll
    for (int s = 0; s < 2; ++s) {
        const int mrow = m0 + s*16 + (lane & 15);
        const int nimg = mrow / SP;
        const int rem  = mrow - nimg * SP;
        const int oh   = rem / HW;
        const int ow   = rem - oh * HW;
        pixA[s] = (long long)nimg * PIX + (long long)oh * HP + ow;
    }

    v8f acc[2][4];
    #pragma unroll
    for (int s = 0; s < 2; ++s)
        #pragma unroll
        for (int t = 0; t < 4; ++t)
            acc[s][t] = v8f{};

    const int col = n0 + (lane & 15);

    #pragma unroll
    for (int kh = 0; kh < 3; ++kh) {
        #pragma unroll
        for (int kw = 0; kw < 3; ++kw) {
            const unsigned char* wtap = wp + (long long)((kh*3 + kw) * OC) * C;
            #pragma unroll
            for (int kb = 0; kb < 2; ++kb) {
                union { v16i v; unsigned long long q[8]; } A[2];
                #pragma unroll
                for (int s = 0; s < 2; ++s) {
                    const unsigned char* ab =
                        xp + (pixA[s] + kh*HP + kw) * C + kb*128 + half*16;
                    #pragma unroll
                    for (int j = 0; j < 4; ++j) {
                        A[s].q[j]     = *(const unsigned long long*)(ab + (j>>1)*32 + (j&1)*8);
                        A[s].q[j + 4] = *(const unsigned long long*)(ab + 64 + (j>>1)*32 + (j&1)*8);
                    }
                }
                #pragma unroll
                for (int t = 0; t < 4; ++t) {
                    union { v16i v; U128 x[4]; } B;
                    const unsigned char* bb =
                        wtap + (long long)(col + t*16) * C + kb*128 + half*16;
                    #pragma unroll
                    for (int q = 0; q < 4; ++q)
                        B.x[q] = *(const U128*)(bb + q*32);
                    acc[0][t] = __builtin_amdgcn_wmma_f32_16x16x128_fp8_fp8(
                        A[0].v, B.v, (short)0, acc[0][t], false, false);
                    acc[1][t] = __builtin_amdgcn_wmma_f32_16x16x128_fp8_fp8(
                        A[1].v, B.v, (short)0, acc[1][t], false, false);
                }
            }
        }
    }

    const int occ = lane & 15;
    #pragma unroll
    for (int s = 0; s < 2; ++s) {
        const int mb     = m0 + s*16;
        const int nI     = mb / SP;
        const int spBase = mb - nI * SP + 8 * half;
        #pragma unroll
        for (int t = 0; t < 4; ++t) {
            const int oc = n0 + t*16 + occ;
            float* ob = out + ((long long)nI * OC + oc) * SP + spBase;
            #pragma unroll
            for (int r = 0; r < 8; ++r)
                ob[r] = acc[s][t][r];
        }
    }
}

extern "C" void kernel_launch(void* const* d_in, const int* in_sizes, int n_in,
                              void* d_out, int out_size, void* d_ws, size_t ws_size,
                              hipStream_t stream) {
    const float* x   = (const float*)d_in[0];
    const float* wgt = (const float*)d_in[1];
    float* out = (float*)d_out;

    unsigned char* xp = (unsigned char*)d_ws;
    unsigned char* wp = xp + XP_BYTES;

    hipMemsetAsync(xp, 0, (size_t)XP_BYTES, stream);

    {
        long long tot = (long long)NB * C * SP;
        int blocks = (int)((tot + 255) / 256);
        pack_x<<<blocks, 256, 0, stream>>>(x, xp);
    }
    pack_w<<<(9*OC*C + 255)/256, 256, 0, stream>>>(wgt, wp);

    bconv_wmma<<<M_TOT/64, 256, 0, stream>>>(xp, wp, out);
}
